// Attention_45148696216131
// MI455X (gfx1250) — matrix operand order repaired and checked by reading, not compiled, not run on hardware
//
#include <hip/hip_runtime.h>
#include <hip/hip_bf16.h>

typedef __bf16 bf16;
typedef __attribute__((ext_vector_type(16))) __bf16 v16bf;
typedef __attribute__((ext_vector_type(8)))  __bf16 v8bf;
typedef __attribute__((ext_vector_type(4)))  __bf16 v4bf;
typedef __attribute__((ext_vector_type(8)))  float  v8f;

#define DIM    1024
#define HEADS  16
#define HD     64
#define SEQ    2048
#define BATCH  4
#define ROWS   (BATCH*SEQ)   // 8192

#define WMMA_BF16(a,b,c) __builtin_amdgcn_wmma_f32_16x16x32_bf16(false,(a),false,(b),(short)0,(c),false,false)

// Async 16B copy global -> LDS (tracked by ASYNCcnt), per CDNA5 ISA
#define ASYNC_CP16(ldsaddr, gptr) \
    asm volatile("global_load_async_to_lds_b128 %0, %1, off" :: "v"(ldsaddr), "v"(gptr) : "memory")
#define ASYNC_WAIT() asm volatile("s_wait_asynccnt 0x0" ::: "memory")

// Pin 8 fragments simultaneously live so all ds_loads issue as one burst with a
// single s_wait_dscnt, instead of load->wait->wmma serialization.
static __device__ __forceinline__ void pin8(v16bf& b0, v16bf& b1, v16bf& b2, v16bf& b3,
                                            v16bf& b4, v16bf& b5, v16bf& b6, v16bf& b7) {
    asm volatile("" : "+v"(b0), "+v"(b1), "+v"(b2), "+v"(b3),
                      "+v"(b4), "+v"(b5), "+v"(b6), "+v"(b7));
}

// ---------------- fragment loaders (per CDNA5 ISA 7.12.2) ----------
// A 16x32 bf16, row-major: lane m=lane&15, half=lane>>4, elem e -> K=(e%8)+half*8+(e/8)*16
static __device__ __forceinline__ v16bf load_a16x32(const bf16* base, int ld, int lane) {
    const int m = lane & 15, h = lane >> 4;
    const bf16* p = base + m * ld + h * 8;
    v8bf lo = *(const v8bf*)p;
    v8bf hi = *(const v8bf*)(p + 16);
    v16bf a;
#pragma unroll
    for (int e = 0; e < 8; ++e) { a[e] = lo[e]; a[e + 8] = hi[e]; }
    return a;
}

// B 32x16 bf16 from transposed storage Bt[n][k]: lane n=lane&15, elem e -> K=(e%8)+half*8+(e/8)*16
static __device__ __forceinline__ v16bf load_b32x16T(const bf16* baseT, int ldt, int lane) {
    const int n = lane & 15, h = lane >> 4;
    const bf16* p = baseT + n * ldt + h * 8;
    v8bf lo = *(const v8bf*)p;
    v8bf hi = *(const v8bf*)(p + 16);
    v16bf b;
#pragma unroll
    for (int e = 0; e < 8; ++e) { b[e] = lo[e]; b[e + 8] = hi[e]; }
    return b;
}
// C/D layout: lane n=lane&15, elem r -> row m = r + 8*(lane>>4)

// ---------------- kernels ----------------------------------------------------
__global__ void cvt_f32_bf16(const float* __restrict__ in, bf16* __restrict__ out, int n) {
    int i = blockIdx.x * blockDim.x + threadIdx.x;
    int stride = gridDim.x * blockDim.x;
    for (; i < n; i += stride) out[i] = (bf16)in[i];
}

// Shared 64x128-tile GEMM mainloop, K-chunk = 64: acc[8] += A[64xK] @ W[Kx128]
static __device__ __forceinline__ void gemm64x128_mainloop(
    const bf16* __restrict__ Xg, const bf16* __restrict__ Wg, int ldw,
    int m0, int n0, bf16* Ash /*64x64*/, bf16* WT /*128x64 (n-major)*/, v8f acc[8])
{
    const int t = threadIdx.x;
    const int lane = t & 31, wave = t >> 5;
    const unsigned ldsA = (unsigned)(uintptr_t)Ash;
    const int arow = t >> 3, aseg = t & 7;            // A tile 64x64: 4 passes of 16 rows x 8 segs
    const int wk = (t & 7) * 8, wn = (t >> 3) * 8;    // W tile 64x128: 8 rows x 8 cols per thread

    for (int kk = 0; kk < DIM; kk += 64) {
        __syncthreads();
        // async stage A tile (64x64 bf16 = 8KB), 4x16B per thread
        {
            const bf16* g = Xg + (size_t)(m0 + arow) * DIM + kk + aseg * 8;
            unsigned l = ldsA + arow * 128 + aseg * 16;
#pragma unroll
            for (int p = 0; p < 4; ++p) {
                ASYNC_CP16(l, g);
                g += (size_t)16 * DIM;
                l += 16 * 128;
            }
        }
        // stage W tile (64x128) transposed, packed b64 stores
        const bf16* wp = Wg + (size_t)(kk + wk) * ldw + n0 + wn;
        {
            v8bf w[8];
#pragma unroll
            for (int rr = 0; rr < 8; ++rr) w[rr] = *(const v8bf*)(wp + (size_t)rr * ldw);
#pragma unroll
            for (int c = 0; c < 8; ++c) {
                v4bf p0, p1;
                p0[0] = w[0][c]; p0[1] = w[1][c]; p0[2] = w[2][c]; p0[3] = w[3][c];
                p1[0] = w[4][c]; p1[1] = w[5][c]; p1[2] = w[6][c]; p1[3] = w[7][c];
                *(v4bf*)(WT + (wn + c) * 64 + wk)     = p0;
                *(v4bf*)(WT + (wn + c) * 64 + wk + 4) = p1;
            }
        }
        if (kk + 64 < DIM) __builtin_prefetch(wp + (size_t)64 * ldw, 0, 1);

        ASYNC_WAIT();
        __syncthreads();

#pragma unroll
        for (int half = 0; half < 2; ++half) {
            v16bf a  = load_a16x32(Ash + wave * 16 * 64 + half * 32, 64, lane);
            v16bf b0 = load_b32x16T(WT + 0 * 16 * 64 + half * 32, 64, lane);
            v16bf b1 = load_b32x16T(WT + 1 * 16 * 64 + half * 32, 64, lane);
            v16bf b2 = load_b32x16T(WT + 2 * 16 * 64 + half * 32, 64, lane);
            v16bf b3 = load_b32x16T(WT + 3 * 16 * 64 + half * 32, 64, lane);
            v16bf b4 = load_b32x16T(WT + 4 * 16 * 64 + half * 32, 64, lane);
            v16bf b5 = load_b32x16T(WT + 5 * 16 * 64 + half * 32, 64, lane);
            v16bf b6 = load_b32x16T(WT + 6 * 16 * 64 + half * 32, 64, lane);
            v16bf b7 = load_b32x16T(WT + 7 * 16 * 64 + half * 32, 64, lane);
            pin8(b0, b1, b2, b3, b4, b5, b6, b7);
            acc[0] = WMMA_BF16(a, b0, acc[0]);
            acc[1] = WMMA_BF16(a, b1, acc[1]);
            acc[2] = WMMA_BF16(a, b2, acc[2]);
            acc[3] = WMMA_BF16(a, b3, acc[3]);
            acc[4] = WMMA_BF16(a, b4, acc[4]);
            acc[5] = WMMA_BF16(a, b5, acc[5]);
            acc[6] = WMMA_BF16(a, b6, acc[6]);
            acc[7] = WMMA_BF16(a, b7, acc[7]);
        }
    }
}

// QKV projection -> Q,K,V in [B,H,S,HD] bf16 (Q pre-scaled by 1/sqrt(HD))
__global__ __launch_bounds__(128) void qkv_gemm_kernel(
    const bf16* __restrict__ X, const bf16* __restrict__ W,
    const float* __restrict__ bias,
    bf16* __restrict__ Q, bf16* __restrict__ K, bf16* __restrict__ V)
{
    __shared__ __align__(16) bf16 Ash[64 * 64];
    __shared__ __align__(16) bf16 WT[128 * 64];
    const int lane = threadIdx.x & 31;
    const int wave = threadIdx.x >> 5;
    const int n0 = blockIdx.x * 128;
    const int m0 = blockIdx.y * 64;
    v8f acc[8] = {};
    gemm64x128_mainloop(X, W, 3 * DIM, m0, n0, Ash, WT, acc);

    const int n = lane & 15, hh = lane >> 4;
    const float qscale = 0.125f;  // 1/sqrt(HD)
#pragma unroll
    for (int nt = 0; nt < 8; ++nt) {
        int col = n0 + nt * 16 + n;
        float bval = bias[col];
        int head = col / (3 * HD);
        int w3 = col % (3 * HD);
#pragma unroll
        for (int r = 0; r < 8; ++r) {
            int row = m0 + wave * 16 + r + 8 * hh;
            int bidx = row / SEQ, s = row % SEQ;
            float val = acc[nt][r] + bval;
            size_t dst = (((size_t)bidx * HEADS + head) * SEQ + s) * HD;
            if (w3 < HD)        Q[dst + w3]          = (bf16)(val * qscale);
            else if (w3 < 2*HD) K[dst + (w3 - HD)]   = (bf16)val;
            else                V[dst + (w3 - 2*HD)] = (bf16)val;
        }
    }
}

// Flash attention: per block 64 query rows (4 waves x 16), loop over 64-key chunks.
__global__ __launch_bounds__(128) void attn_kernel(
    const bf16* __restrict__ Q, const bf16* __restrict__ K, const bf16* __restrict__ V,
    bf16* __restrict__ O)  // [B,S,H,HD]
{
    __shared__ __align__(16) bf16 Ksh[64 * 64];     // row-major chunk (== transposed-B for QK^T)
    __shared__ __align__(16) bf16 VT[64 * 64];      // VT[d][k]
    __shared__ __align__(16) bf16 Psh[4 * 16 * 64]; // per-wave P tiles
    const int lane = threadIdx.x & 31;
    const int wave = threadIdx.x >> 5;
    const int bh = blockIdx.y;
    const int b = bh / HEADS, h = bh % HEADS;
    const int q0 = blockIdx.x * 64 + wave * 16;
    const bf16* Qp = Q + (size_t)bh * SEQ * HD;
    const bf16* Kp = K + (size_t)bh * SEQ * HD;
    const bf16* Vp = V + (size_t)bh * SEQ * HD;

    v16bf aq0 = load_a16x32(Qp + (size_t)q0 * HD, HD, lane);
    v16bf aq1 = load_a16x32(Qp + (size_t)q0 * HD + 32, HD, lane);
    v8f o0 = {}, o1 = {}, o2 = {}, o3 = {};
    float rm[8], lsum[8];
#pragma unroll
    for (int r = 0; r < 8; ++r) { rm[r] = -1e30f; lsum[r] = 0.f; }

    const int t = threadIdx.x;
    const unsigned ldsK = (unsigned)(uintptr_t)Ksh;
    const int krow = t >> 3, kseg = t & 7;           // K chunk 64x64: 4 passes of 16 rows
    const int vk = (t & 7) * 8, vn = (t >> 3) * 4;   // V tile: 8 rows x 4 cols per thread
    bf16* Pw = Psh + wave * 16 * 64;
    const int n = lane & 15, hh = lane >> 4;

    for (int j0 = 0; j0 < SEQ; j0 += 64) {
        __syncthreads();
        // async stage K chunk (64x64 bf16 = 8KB), 4x16B per thread
        {
            const bf16* kg = Kp + (size_t)(j0 + krow) * HD + kseg * 8;
            unsigned kl = ldsK + krow * 128 + kseg * 16;
#pragma unroll
            for (int p = 0; p < 4; ++p) {
                ASYNC_CP16(kl, kg);
                kg += (size_t)16 * HD;
                kl += 16 * 128;
            }
        }
        // stage V chunk (64x64) transposed, packed b64 stores
        const bf16* vgp = Vp + (size_t)(j0 + vk) * HD + vn;
        {
            v4bf x[8];
#pragma unroll
            for (int rr = 0; rr < 8; ++rr) x[rr] = *(const v4bf*)(vgp + (size_t)rr * HD);
#pragma unroll
            for (int c = 0; c < 4; ++c) {
                v4bf p0, p1;
                p0[0] = x[0][c]; p0[1] = x[1][c]; p0[2] = x[2][c]; p0[3] = x[3][c];
                p1[0] = x[4][c]; p1[1] = x[5][c]; p1[2] = x[6][c]; p1[3] = x[7][c];
                *(v4bf*)(VT + (vn + c) * 64 + vk)     = p0;
                *(v4bf*)(VT + (vn + c) * 64 + vk + 4) = p1;
            }
        }
        if (j0 + 64 < SEQ) {
            __builtin_prefetch(Kp + (size_t)(j0 + 64 + krow) * HD + kseg * 8, 0, 1);
            __builtin_prefetch(vgp + (size_t)64 * HD, 0, 1);
        }
        ASYNC_WAIT();
        __syncthreads();

        // logits for 64 keys: 4 C tiles, contraction over HD=64 (2 WMMA each)
        v8f c0 = {}, c1 = {}, c2 = {}, c3 = {};
        {
            v16bf k0 = load_b32x16T(Ksh + 0 * 16 * 64, 64, lane);
            v16bf k1 = load_b32x16T(Ksh + 1 * 16 * 64, 64, lane);
            v16bf k2 = load_b32x16T(Ksh + 2 * 16 * 64, 64, lane);
            v16bf k3 = load_b32x16T(Ksh + 3 * 16 * 64, 64, lane);
            v16bf k4 = load_b32x16T(Ksh + 0 * 16 * 64 + 32, 64, lane);
            v16bf k5 = load_b32x16T(Ksh + 1 * 16 * 64 + 32, 64, lane);
            v16bf k6 = load_b32x16T(Ksh + 2 * 16 * 64 + 32, 64, lane);
            v16bf k7 = load_b32x16T(Ksh + 3 * 16 * 64 + 32, 64, lane);
            pin8(k0, k1, k2, k3, k4, k5, k6, k7);
            c0 = WMMA_BF16(aq0, k0, c0);
            c1 = WMMA_BF16(aq0, k1, c1);
            c2 = WMMA_BF16(aq0, k2, c2);
            c3 = WMMA_BF16(aq0, k3, c3);
            c0 = WMMA_BF16(aq1, k4, c0);
            c1 = WMMA_BF16(aq1, k5, c1);
            c2 = WMMA_BF16(aq1, k6, c2);
            c3 = WMMA_BF16(aq1, k7, c3);
        }

        // online softmax over 64 keys (row = r + 8*hh spans one 16-lane half-group)
#pragma unroll
        for (int r = 0; r < 8; ++r) {
            float mx = fmaxf(fmaxf(c0[r], c1[r]), fmaxf(c2[r], c3[r]));
#pragma unroll
            for (int msk = 1; msk < 16; msk <<= 1) mx = fmaxf(mx, __shfl_xor(mx, msk, 32));
            float nm = fmaxf(rm[r], mx);
            float sc = __expf(rm[r] - nm);
            rm[r] = nm;
            float p0 = __expf(c0[r] - nm);
            float p1 = __expf(c1[r] - nm);
            float p2 = __expf(c2[r] - nm);
            float p3 = __expf(c3[r] - nm);
            float ps = (p0 + p1) + (p2 + p3);
#pragma unroll
            for (int msk = 1; msk < 16; msk <<= 1) ps += __shfl_xor(ps, msk, 32);
            lsum[r] = lsum[r] * sc + ps;
            o0[r] *= sc; o1[r] *= sc; o2[r] *= sc; o3[r] *= sc;
            int m = r + 8 * hh;
            Pw[m * 64 + n]      = (bf16)p0;
            Pw[m * 64 + 16 + n] = (bf16)p1;
            Pw[m * 64 + 32 + n] = (bf16)p2;
            Pw[m * 64 + 48 + n] = (bf16)p3;
        }

        // O += P[16x64] @ V[64x64]
        {
            v16bf ap0 = load_a16x32(Pw, 64, lane);
            v16bf ap1 = load_a16x32(Pw + 32, 64, lane);
            v16bf w0 = load_b32x16T(VT + 0 * 16 * 64, 64, lane);
            v16bf w1 = load_b32x16T(VT + 1 * 16 * 64, 64, lane);
            v16bf w2 = load_b32x16T(VT + 2 * 16 * 64, 64, lane);
            v16bf w3 = load_b32x16T(VT + 3 * 16 * 64, 64, lane);
            v16bf w4 = load_b32x16T(VT + 0 * 16 * 64 + 32, 64, lane);
            v16bf w5 = load_b32x16T(VT + 1 * 16 * 64 + 32, 64, lane);
            v16bf w6 = load_b32x16T(VT + 2 * 16 * 64 + 32, 64, lane);
            v16bf w7 = load_b32x16T(VT + 3 * 16 * 64 + 32, 64, lane);
            pin8(w0, w1, w2, w3, w4, w5, w6, w7);
            o0 = WMMA_BF16(ap0, w0, o0);
            o1 = WMMA_BF16(ap0, w1, o1);
            o2 = WMMA_BF16(ap0, w2, o2);
            o3 = WMMA_BF16(ap0, w3, o3);
            o0 = WMMA_BF16(ap1, w4, o0);
            o1 = WMMA_BF16(ap1, w5, o1);
            o2 = WMMA_BF16(ap1, w6, o2);
            o3 = WMMA_BF16(ap1, w7, o3);
        }
    }

#pragma unroll
    for (int r = 0; r < 8; ++r) {
        float inv = 1.0f / lsum[r];
        int s = q0 + r + 8 * hh;
        size_t base = (((size_t)b * SEQ + s) * HEADS + h) * HD + n;
        O[base + 0]  = (bf16)(o0[r] * inv);
        O[base + 16] = (bf16)(o1[r] * inv);
        O[base + 32] = (bf16)(o2[r] * inv);
        O[base + 48] = (bf16)(o3[r] * inv);
    }
}

// Output projection: vals [8192,1024] @ W_out [1024,1024] + bias -> f32 out
__global__ __launch_bounds__(128) void out_gemm_kernel(
    const bf16* __restrict__ A, const bf16* __restrict__ W,
    const float* __restrict__ bias, float* __restrict__ Out)
{
    __shared__ __align__(16) bf16 Ash[64 * 64];
    __shared__ __align__(16) bf16 WT[128 * 64];
    const int lane = threadIdx.x & 31;
    const int wave = threadIdx.x >> 5;
    const int n0 = blockIdx.x * 128;
    const int m0 = blockIdx.y * 64;
    v8f acc[8] = {};
    gemm64x128_mainloop(A, W, DIM, m0, n0, Ash, WT, acc);

    const int n = lane & 15, hh = lane >> 4;
#pragma unroll
    for (int nt = 0; nt < 8; ++nt) {
        int col = n0 + nt * 16 + n;
        float bval = bias[col];
#pragma unroll
        for (int r = 0; r < 8; ++r) {
            int row = m0 + wave * 16 + r + 8 * hh;
            Out[(size_t)row * DIM + col] = acc[nt][r] + bval;
        }
    }
}

// ---------------- launch -----------------------------------------------------
extern "C" void kernel_launch(void* const* d_in, const int* in_sizes, int n_in,
                              void* d_out, int out_size, void* d_ws, size_t ws_size,
                              hipStream_t stream) {
    const float* x     = (const float*)d_in[0];
    const float* W_qkv = (const float*)d_in[1];
    const float* b_qkv = (const float*)d_in[2];
    const float* W_out = (const float*)d_in[3];
    const float* b_out = (const float*)d_in[4];

    char* ws = (char*)d_ws;
    size_t off = 0;
    bf16* xb    = (bf16*)(ws + off); off += (size_t)ROWS * DIM * sizeof(bf16);
    bf16* wqkvb = (bf16*)(ws + off); off += (size_t)DIM * 3 * DIM * sizeof(bf16);
    bf16* woutb = (bf16*)(ws + off); off += (size_t)DIM * DIM * sizeof(bf16);
    bf16* Qb    = (bf16*)(ws + off); off += (size_t)ROWS * DIM * sizeof(bf16);
    bf16* Kb    = (bf16*)(ws + off); off += (size_t)ROWS * DIM * sizeof(bf16);
    bf16* Vb    = (bf16*)(ws + off); off += (size_t)ROWS * DIM * sizeof(bf16);
    bf16* vals  = (bf16*)(ws + off); off += (size_t)ROWS * DIM * sizeof(bf16);

    cvt_f32_bf16<<<4096, 256, 0, stream>>>(x, xb, ROWS * DIM);
    cvt_f32_bf16<<<4096, 256, 0, stream>>>(W_qkv, wqkvb, DIM * 3 * DIM);
    cvt_f32_bf16<<<2048, 256, 0, stream>>>(W_out, woutb, DIM * DIM);

    qkv_gemm_kernel<<<dim3(3 * DIM / 128, ROWS / 64), 128, 0, stream>>>(
        xb, wqkvb, b_qkv, Qb, Kb, Vb);

    attn_kernel<<<dim3(SEQ / 64, BATCH * HEADS), 128, 0, stream>>>(
        Qb, Kb, Vb, vals);

    out_gemm_kernel<<<dim3(DIM / 128, ROWS / 64), 128, 0, stream>>>(
        vals, woutb, b_out, (float*)d_out);
}
